// Model_78469052498039
// MI455X (gfx1250) — matrix operand order repaired and checked by reading, not compiled, not run on hardware
//
#include <hip/hip_runtime.h>
#include <math.h>

#define NODES 6000
#define EDGES 192000
#define DIM   512
#define OUTD  40
#define KCB   4096

typedef __attribute__((ext_vector_type(16))) _Float16 v16h;
typedef __attribute__((ext_vector_type(8)))  _Float16 v8h;
typedef __attribute__((ext_vector_type(8)))  float    v8f;

// ---------- order-preserving float <-> uint for atomic min/max ----------
__device__ __forceinline__ unsigned fkey(float f){
  unsigned u = __float_as_uint(f);
  return (u & 0x80000000u) ? ~u : (u | 0x80000000u);
}
__device__ __forceinline__ float funkey(unsigned k){
  unsigned u = (k & 0x80000000u) ? (k & 0x7fffffffu) : ~k;
  return __uint_as_float(u);
}

// ---------- WMMA fragment helpers (CDNA5 16x16x32 f16 layouts) ----------
// A 16x32: lane=(half*16+r) holds row r; elems 0..7 -> K=half*8+e, 8..15 -> K=16+half*8+(e-8)
// B 32x16: lane=(half*16+n) holds col n; same K order as A (loaded from Bt rows via load_afrag)
__device__ __forceinline__ v16h load_afrag(const _Float16* __restrict__ arow, int kk, int half){
  v8h s0 = *(const v8h*)(arow + kk + half*8);
  v8h s1 = *(const v8h*)(arow + kk + 16 + half*8);
  v16h af;
#pragma unroll
  for (int i = 0; i < 8; ++i){ af[i] = s0[i]; af[i+8] = s1[i]; }
  return af;
}

// ---------- 2x2-tile (32x32) WMMA core: 4 WMMAs per 8 fragment loads ----------
template<bool R2, bool C2>
__device__ __forceinline__ void wmma_core_2x2(const _Float16* __restrict__ A,
                                              const _Float16* __restrict__ Bt,
                                              int K, int row0, int col0, int lane,
                                              v8f& a00, v8f& a01, v8f& a10, v8f& a11){
  const int half = lane >> 4, r = lane & 15;
  const _Float16* ar0 = A  + (size_t)(row0 + r) * K;
  const _Float16* ar1 = A  + (size_t)(row0 + 16 + r) * K;
  const _Float16* br0 = Bt + (size_t)(col0 + r) * K;
  const _Float16* br1 = Bt + (size_t)(col0 + 16 + r) * K;
  for (int kk = 0; kk < K; kk += 32){
    v16h af0 = load_afrag(ar0, kk, half);
    v16h bf0 = load_afrag(br0, kk, half);
    v16h af1, bf1;
    if (R2) af1 = load_afrag(ar1, kk, half);
    if (C2) bf1 = load_afrag(br1, kk, half);
    a00 = __builtin_amdgcn_wmma_f32_16x16x32_f16(false, af0, false, bf0, (short)0, a00, false, false);
    if (C2)       a01 = __builtin_amdgcn_wmma_f32_16x16x32_f16(false, af0, false, bf1, (short)0, a01, false, false);
    if (R2)       a10 = __builtin_amdgcn_wmma_f32_16x16x32_f16(false, af1, false, bf0, (short)0, a10, false, false);
    if (R2 && C2) a11 = __builtin_amdgcn_wmma_f32_16x16x32_f16(false, af1, false, bf1, (short)0, a11, false, false);
  }
}

__device__ __forceinline__ void wmma_dispatch_2x2(const _Float16* A, const _Float16* Bt,
                                                  int K, int row0, int col0, int lane,
                                                  bool R2, bool C2,
                                                  v8f& a00, v8f& a01, v8f& a10, v8f& a11){
  if (R2){
    if (C2) wmma_core_2x2<true, true >(A, Bt, K, row0, col0, lane, a00, a01, a10, a11);
    else    wmma_core_2x2<true, false>(A, Bt, K, row0, col0, lane, a00, a01, a10, a11);
  } else {
    if (C2) wmma_core_2x2<false, true >(A, Bt, K, row0, col0, lane, a00, a01, a10, a11);
    else    wmma_core_2x2<false, false>(A, Bt, K, row0, col0, lane, a00, a01, a10, a11);
  }
}

// ---------- degree counting ----------
__global__ void vq_deg_count(const int* __restrict__ src, const int* __restrict__ dst,
                             unsigned* co, unsigned* ci){
  int i = blockIdx.x * blockDim.x + threadIdx.x;
  if (i >= EDGES) return;
  atomicAdd(&co[src[i]], 1u);
  atomicAdd(&ci[dst[i]], 1u);
}
__global__ void vq_deg_final(const unsigned* co, const unsigned* ci,
                             float* ro, float* ri){
  int i = blockIdx.x * blockDim.x + threadIdx.x;
  if (i >= NODES) return;
  ro[i] = rsqrtf(fmaxf((float)co[i], 1.0f));
  ri[i] = rsqrtf(fmaxf((float)ci[i], 1.0f));
}

// ---------- edge scatter: agg[dst] += x[src] * r_out[src] ----------
__global__ void vq_scatter(const float* __restrict__ X, const float* __restrict__ ro,
                           const int* __restrict__ src, const int* __restrict__ dst,
                           float* agg){
  int e = blockIdx.x;
  int s = src[e], t = dst[e];
  float r = ro[s];
  const float4* xs = (const float4*)(X + (size_t)s * DIM);
  float* ag = agg + (size_t)t * DIM;
  float4 v = xs[threadIdx.x];            // blockDim.x == 128, 128*4 = 512
  int d = threadIdx.x * 4;
  atomicAdd(&ag[d+0], v.x * r);
  atomicAdd(&ag[d+1], v.y * r);
  atomicAdd(&ag[d+2], v.z * r);
  atomicAdd(&ag[d+3], v.w * r);
}

// ---------- y_h[n,d] = f16(x[n,d] * rr[n]) ----------
__global__ void vq_scale_convert(const float* __restrict__ X, const float* __restrict__ rr,
                                 _Float16* Y){
  size_t idx = (size_t)blockIdx.x * blockDim.x + threadIdx.x;
  if (idx >= (size_t)NODES * DIM) return;
  int n = (int)(idx / DIM);
  Y[idx] = (_Float16)(X[idx] * rr[n]);
}

__global__ void vq_convert_h(const float* __restrict__ X, _Float16* Y, size_t n){
  size_t idx = (size_t)blockIdx.x * blockDim.x + threadIdx.x;
  if (idx < n) Y[idx] = (_Float16)X[idx];
}

// ---------- Wt[n,k] = f16(W[k,n]) ----------
__global__ void vq_transpose_h(const float* __restrict__ W, _Float16* Wt, int K, int Nn){
  size_t idx = (size_t)blockIdx.x * blockDim.x + threadIdx.x;
  if (idx >= (size_t)K * Nn) return;
  int k = (int)(idx / Nn), n = (int)(idx % Nn);
  Wt[(size_t)n * K + k] = (_Float16)W[idx];
}

// ---------- row-wise l2 normalize (rsqrt(ss + 1e-12)) ----------
__global__ void vq_l2norm(const float* __restrict__ X, float* Yf, _Float16* Yh){
  int row = blockIdx.x;
  __shared__ float red[256];
  const float* xr = X + (size_t)row * DIM;
  float s = 0.f;
  for (int j = threadIdx.x; j < DIM; j += 256){ float v = xr[j]; s += v * v; }
  red[threadIdx.x] = s; __syncthreads();
  for (int st = 128; st > 0; st >>= 1){
    if (threadIdx.x < st) red[threadIdx.x] += red[threadIdx.x + st];
    __syncthreads();
  }
  float r = rsqrtf(red[0] + 1e-12f);
  for (int j = threadIdx.x; j < DIM; j += 256){
    float v = xr[j] * r;
    if (Yf) Yf[(size_t)row * DIM + j] = v;
    Yh[(size_t)row * DIM + j] = (_Float16)v;
  }
}

// ---------- WMMA GEMM (32x32 per wave): C = act(A[M,K] * Bt[Nn,K]^T + bias) ----------
__device__ __forceinline__ void store_tile(float* __restrict__ C, int ldc,
                                           const float* __restrict__ bias, int relu,
                                           const v8f& acc, int rowb, int colb,
                                           int half, int r){
  int col = colb + r;
  float bv = bias ? bias[col] : 0.f;
#pragma unroll
  for (int v = 0; v < 8; ++v){
    int row = rowb + half * 8 + v;
    float x = acc[v] + bv;
    if (relu) x = fmaxf(x, 0.f);
    C[(size_t)row * ldc + col] = x;
  }
}

__global__ void vq_gemm_wmma(const _Float16* __restrict__ A, const _Float16* __restrict__ Bt,
                             const float* __restrict__ bias, float* __restrict__ C,
                             int ldc, int M, int Nn, int K, int relu){
  int wave = threadIdx.x >> 5, lane = threadIdx.x & 31;
  int tn = blockIdx.x * (blockDim.x >> 5) + wave;   // wave-uniform
  int tm = blockIdx.y;
  int row0 = tm * 32, col0 = tn * 32;
  if (row0 >= M || col0 >= Nn) return;              // wave-uniform exit
  bool R2 = (row0 + 16) < M;
  bool C2 = (col0 + 16) < Nn;
  v8f a00 = {}, a01 = {}, a10 = {}, a11 = {};
  wmma_dispatch_2x2(A, Bt, K, row0, col0, lane, R2, C2, a00, a01, a10, a11);
  int half = lane >> 4, r = lane & 15;
  store_tile(C, ldc, bias, relu, a00, row0, col0, half, r);
  if (C2)       store_tile(C, ldc, bias, relu, a01, row0,      col0 + 16, half, r);
  if (R2)       store_tile(C, ldc, bias, relu, a10, row0 + 16, col0,      half, r);
  if (R2 && C2) store_tile(C, ldc, bias, relu, a11, row0 + 16, col0 + 16, half, r);
}

// ---------- argmax per row (first-max tie rule) ----------
__global__ void vq_argmax(const float* __restrict__ Dist, int* ind){
  int row = blockIdx.x;
  __shared__ float bv[256]; __shared__ int bi[256];
  float best = -3.4e38f; int bidx = 0;
  const float* dr = Dist + (size_t)row * KCB;
  for (int j = threadIdx.x; j < KCB; j += 256){
    float v = dr[j];
    if (v > best){ best = v; bidx = j; }
  }
  bv[threadIdx.x] = best; bi[threadIdx.x] = bidx; __syncthreads();
  for (int st = 128; st > 0; st >>= 1){
    if (threadIdx.x < st){
      float ov = bv[threadIdx.x + st]; int oi = bi[threadIdx.x + st];
      if (ov > bv[threadIdx.x] || (ov == bv[threadIdx.x] && oi < bi[threadIdx.x])){
        bv[threadIdx.x] = ov; bi[threadIdx.x] = oi;
      }
    }
    __syncthreads();
  }
  if (threadIdx.x == 0) ind[row] = bi[0];
}

// ---------- q = en[ind], plus f16 copy ----------
__global__ void vq_gather(const float* __restrict__ En, const int* __restrict__ ind,
                          float* Q, _Float16* Qh){
  size_t idx = (size_t)blockIdx.x * blockDim.x + threadIdx.x;
  if (idx >= (size_t)NODES * DIM) return;
  int n = (int)(idx / DIM), d = (int)(idx % DIM);
  float v = En[(size_t)ind[n] * DIM + d];
  Q[idx] = v; Qh[idx] = (_Float16)v;
}

// ---------- sum (a-b)^2 into double accumulator ----------
__global__ void vq_mse(const float* __restrict__ A, const float* __restrict__ B,
                       size_t n, double* acc){
  __shared__ double sh[256];
  double s = 0.0;
  for (size_t i = (size_t)blockIdx.x * blockDim.x + threadIdx.x; i < n;
       i += (size_t)gridDim.x * blockDim.x){
    float d = A[i] - B[i]; s += (double)d * (double)d;
  }
  sh[threadIdx.x] = s; __syncthreads();
  for (int st = 128; st > 0; st >>= 1){
    if (threadIdx.x < st) sh[threadIdx.x] += sh[threadIdx.x + st];
    __syncthreads();
  }
  if (threadIdx.x == 0) atomicAdd(acc, sh[0]);
}

// ---------- adj_q pass 1: global min/max via 32x32 WMMA jobs ----------
__global__ void vq_adjq_minmax(const _Float16* __restrict__ Qh, unsigned* mnk, unsigned* mxk){
  int wave = threadIdx.x >> 5, lane = threadIdx.x & 31;
  const int tiles2 = (NODES + 31) / 32;             // 188 (last ragged)
  int tid = blockIdx.x * (blockDim.x >> 5) + wave;
  if (tid >= tiles2 * tiles2) return;               // wave-uniform
  int tm = tid / tiles2, tn = tid % tiles2;
  int row0 = tm * 32, col0 = tn * 32;
  bool R2 = (row0 + 16) < NODES, C2 = (col0 + 16) < NODES;
  v8f a00 = {}, a01 = {}, a10 = {}, a11 = {};
  wmma_dispatch_2x2(Qh, Qh, DIM, row0, col0, lane, R2, C2, a00, a01, a10, a11);
  float mn = 3.4e38f, mx = -3.4e38f;
#pragma unroll
  for (int v = 0; v < 8; ++v){ mn = fminf(mn, a00[v]); mx = fmaxf(mx, a00[v]); }
  if (C2){
#pragma unroll
    for (int v = 0; v < 8; ++v){ mn = fminf(mn, a01[v]); mx = fmaxf(mx, a01[v]); }
  }
  if (R2){
#pragma unroll
    for (int v = 0; v < 8; ++v){ mn = fminf(mn, a10[v]); mx = fmaxf(mx, a10[v]); }
  }
  if (R2 && C2){
#pragma unroll
    for (int v = 0; v < 8; ++v){ mn = fminf(mn, a11[v]); mx = fmaxf(mx, a11[v]); }
  }
  for (int o = 16; o > 0; o >>= 1){
    mn = fminf(mn, __shfl_xor(mn, o));
    mx = fmaxf(mx, __shfl_xor(mx, o));
  }
  if (lane == 0){ atomicMin(mnk, fkey(mn)); atomicMax(mxk, fkey(mx)); }
}

__global__ void vq_minmax_final(const unsigned* mnk, const unsigned* mxk,
                                float* mnf, float* invf){
  float mn = funkey(*mnk), mx = funkey(*mxk);
  *mnf = mn;
  float d = mx - mn;
  *invf = (d != 0.f) ? 1.f / d : 0.f;
}

// ---------- adj_q pass 2: sum of normalized^2 via 32x32 WMMA jobs ----------
__global__ void vq_adjq_ssq(const _Float16* __restrict__ Qh, const float* mnp,
                            const float* invp, double* acc){
  int wave = threadIdx.x >> 5, lane = threadIdx.x & 31;
  const int tiles2 = (NODES + 31) / 32;
  int tid = blockIdx.x * (blockDim.x >> 5) + wave;
  __shared__ double sh[8];
  double s = 0.0;
  if (tid < tiles2 * tiles2){                       // wave-uniform
    int tm = tid / tiles2, tn = tid % tiles2;
    int row0 = tm * 32, col0 = tn * 32;
    bool R2 = (row0 + 16) < NODES, C2 = (col0 + 16) < NODES;
    v8f a00 = {}, a01 = {}, a10 = {}, a11 = {};
    wmma_dispatch_2x2(Qh, Qh, DIM, row0, col0, lane, R2, C2, a00, a01, a10, a11);
    float mn = *mnp, inv = *invp;
#pragma unroll
    for (int v = 0; v < 8; ++v){ float t = (a00[v] - mn) * inv; s += (double)t * (double)t; }
    if (C2){
#pragma unroll
      for (int v = 0; v < 8; ++v){ float t = (a01[v] - mn) * inv; s += (double)t * (double)t; }
    }
    if (R2){
#pragma unroll
      for (int v = 0; v < 8; ++v){ float t = (a10[v] - mn) * inv; s += (double)t * (double)t; }
    }
    if (R2 && C2){
#pragma unroll
      for (int v = 0; v < 8; ++v){ float t = (a11[v] - mn) * inv; s += (double)t * (double)t; }
    }
  }
  for (int o = 16; o > 0; o >>= 1) s += __shfl_xor(s, o);
  if (lane == 0) sh[wave] = s;
  __syncthreads();
  if (threadIdx.x == 0){
    double t = 0; int nw = blockDim.x >> 5;
    for (int i = 0; i < nw; ++i) t += sh[i];
    atomicAdd(acc, t);
  }
}

// ---------- byte-packed duplicate edge counts ----------
__global__ void vq_count_edges(const int* __restrict__ src, const int* __restrict__ dst,
                               unsigned* counts){
  int i = blockIdx.x * blockDim.x + threadIdx.x;
  if (i >= EDGES) return;
  size_t cell = (size_t)dst[i] * NODES + src[i];
  atomicAdd(&counts[cell >> 2], 1u << ((cell & 3) * 8));
}

// ---------- per-edge-occurrence term: (c - 2*s) ----------
__global__ void vq_edge_terms(const float* __restrict__ Qe, const int* __restrict__ src,
                              const int* __restrict__ dst, const unsigned* __restrict__ counts,
                              const float* mnp, const float* invp, double* acc){
  int wave = threadIdx.x >> 5, lane = threadIdx.x & 31;
  int e = blockIdx.x * (blockDim.x >> 5) + wave;
  __shared__ double sh[8];
  double contrib = 0.0;
  if (e < EDGES){
    int s = src[e], t = dst[e];
    const float* a = Qe + (size_t)t * DIM;
    const float* b = Qe + (size_t)s * DIM;
    float p = 0.f;
    for (int j = lane; j < DIM; j += 32) p += a[j] * b[j];
    for (int o = 16; o > 0; o >>= 1) p += __shfl_xor(p, o);
    if (lane == 0){
      size_t cell = (size_t)t * NODES + s;
      unsigned c = (counts[cell >> 2] >> ((cell & 3) * 8)) & 0xFFu;
      float sv = (p - *mnp) * (*invp);
      contrib = (double)c - 2.0 * (double)sv;
    }
  }
  if (lane == 0) sh[wave] = contrib;
  __syncthreads();
  if (threadIdx.x == 0){
    double t = 0; int nw = blockDim.x >> 5;
    for (int i = 0; i < nw; ++i) t += sh[i];
    atomicAdd(acc, t);
  }
}

// ---------- final linear: out[n,o] = h2[n,:] @ Wl[:,o] + bl[o] ----------
__global__ void vq_linear_out(const float* __restrict__ H2, const float* __restrict__ Wl,
                              const float* __restrict__ bl, float* __restrict__ Out){
  int idx = blockIdx.x * blockDim.x + threadIdx.x;
  if (idx >= NODES * OUTD) return;
  int n = idx / OUTD, o = idx % OUTD;
  const float* hr = H2 + (size_t)n * DIM;
  float s = bl[o];
  for (int k = 0; k < DIM; ++k) s = fmaf(hr[k], Wl[k * OUTD + o], s);
  Out[idx] = s;
}

// ---------- combine losses ----------
__global__ void vq_loss_final(const double* accCommit, const double* accNode,
                              const double* accS2, const double* accEdge, float* lossOut){
  double nd = *accNode / ((double)NODES * (double)DIM);            // LAMB_NODE = 1
  double cm = 0.25 * (*accCommit) / ((double)NODES * (double)DIM); // COMMIT_W = 0.25
  double e2 = (*accS2 + *accEdge) / ((double)NODES * (double)NODES);
  if (e2 < 0.0) e2 = 0.0;
  *lossOut = (float)(nd + sqrt(e2) + cm);                          // LAMB_EDGE = 1
}

extern "C" void kernel_launch(void* const* d_in, const int* in_sizes, int n_in,
                              void* d_out, int out_size, void* d_ws, size_t ws_size,
                              hipStream_t stream){
  const float* feats    = (const float*)d_in[0];
  const float* W1       = (const float*)d_in[1];
  const float* b1       = (const float*)d_in[2];
  const float* W2       = (const float*)d_in[3];
  const float* b2       = (const float*)d_in[4];
  const float* Wd1      = (const float*)d_in[5];
  const float* bd1      = (const float*)d_in[6];
  const float* Wd2      = (const float*)d_in[7];
  const float* bd2      = (const float*)d_in[8];
  const float* Wl       = (const float*)d_in[9];
  const float* bl       = (const float*)d_in[10];
  const float* codebook = (const float*)d_in[11];
  const int*   src      = (const int*)d_in[12];
  const int*   dst      = (const int*)d_in[13];

  float* outp  = (float*)d_out;                 // [NODES*OUTD]
  float* lossp = outp + NODES * OUTD;           // [1]
  float* distp = lossp + 1;                     // [NODES*KCB]

  // ---- workspace layout (256B aligned) ----
  char* ws = (char*)d_ws; size_t off = 0;
  auto alloc = [&](size_t bytes) -> char* {
    char* p = ws + off; off = (off + bytes + 255) & ~(size_t)255; return p;
  };
  const size_t ND = (size_t)NODES * DIM;
  unsigned* deg_o  = (unsigned*)alloc(NODES * 4);
  unsigned* deg_i  = (unsigned*)alloc(NODES * 4);
  float* r_out     = (float*)alloc(NODES * 4);
  float* r_in      = (float*)alloc(NODES * 4);
  float* agg       = (float*)alloc(ND * 4);
  float* agg2      = (float*)alloc(ND * 4);
  float* h         = (float*)alloc(ND * 4);
  float* q         = (float*)alloc(ND * 4);
  float* q_edge    = (float*)alloc(ND * 4);
  float* q_node    = (float*)alloc(ND * 4);
  float* h2        = (float*)alloc(ND * 4);
  float* en        = (float*)alloc((size_t)KCB * DIM * 4);
  _Float16* en_h   = (_Float16*)alloc((size_t)KCB * DIM * 2);
  _Float16* a1_h   = (_Float16*)alloc(ND * 2);
  _Float16* hn_h   = (_Float16*)alloc(ND * 2);
  _Float16* q_h    = (_Float16*)alloc(ND * 2);
  _Float16* qe_h   = (_Float16*)alloc(ND * 2);
  _Float16* a2_h   = (_Float16*)alloc(ND * 2);
  _Float16* W1t    = (_Float16*)alloc((size_t)DIM * DIM * 2);
  _Float16* W2t    = (_Float16*)alloc((size_t)DIM * DIM * 2);
  _Float16* Wd1t   = (_Float16*)alloc((size_t)DIM * DIM * 2);
  _Float16* Wd2t   = (_Float16*)alloc((size_t)DIM * DIM * 2);
  int* ind         = (int*)alloc(NODES * 4);
  unsigned* counts = (unsigned*)alloc(((size_t)NODES * NODES / 4) * 4);
  double* accs     = (double*)alloc(4 * 8);     // commit, node, s2, edge
  unsigned* mnk    = (unsigned*)alloc(4);
  unsigned* mxk    = (unsigned*)alloc(4);
  float* mnf       = (float*)alloc(4);
  float* invf      = (float*)alloc(4);
  (void)ws_size; (void)in_sizes; (void)n_in; (void)out_size;

  // ---- zero / init accumulators (capture-safe) ----
  hipMemsetAsync(deg_o, 0, NODES * 4, stream);
  hipMemsetAsync(deg_i, 0, NODES * 4, stream);
  hipMemsetAsync(agg, 0, ND * 4, stream);
  hipMemsetAsync(agg2, 0, ND * 4, stream);
  hipMemsetAsync(counts, 0, (size_t)NODES * NODES, stream);
  hipMemsetAsync(accs, 0, 4 * 8, stream);
  hipMemsetAsync(mnk, 0xFF, 4, stream);
  hipMemsetAsync(mxk, 0x00, 4, stream);

  const int T = 256;
  dim3 ndGrid((unsigned)((ND + T - 1) / T));

  // degrees
  vq_deg_count<<<(EDGES + T - 1) / T, T, 0, stream>>>(src, dst, deg_o, deg_i);
  vq_deg_final<<<(NODES + T - 1) / T, T, 0, stream>>>(deg_o, deg_i, r_out, r_in);

  // weight transposes -> f16
  dim3 wGrid((unsigned)(((size_t)DIM * DIM + T - 1) / T));
  vq_transpose_h<<<wGrid, T, 0, stream>>>(W1, W1t, DIM, DIM);
  vq_transpose_h<<<wGrid, T, 0, stream>>>(W2, W2t, DIM, DIM);
  vq_transpose_h<<<wGrid, T, 0, stream>>>(Wd1, Wd1t, DIM, DIM);
  vq_transpose_h<<<wGrid, T, 0, stream>>>(Wd2, Wd2t, DIM, DIM);

  // normalized codebook (f32 + f16)
  vq_l2norm<<<KCB, T, 0, stream>>>(codebook, en, en_h);

  const int mb = (NODES + 31) / 32;             // 188 row-pair strips
  auto gemmGrid = [&](int Nn){ return dim3((unsigned)(((Nn + 31) / 32 + 3) / 4), (unsigned)mb); };

  // GraphConv1: scatter + scale + GEMM(+bias,relu)
  vq_scatter<<<EDGES, 128, 0, stream>>>(feats, r_out, src, dst, agg);
  vq_scale_convert<<<ndGrid, T, 0, stream>>>(agg, r_in, a1_h);
  vq_gemm_wmma<<<gemmGrid(DIM), 128, 0, stream>>>(a1_h, W1t, b1, h, DIM, NODES, DIM, DIM, 1);

  // hn (f16 only), dist = hn @ en^T  -> directly into d_out
  vq_l2norm<<<NODES, T, 0, stream>>>(h, nullptr, hn_h);
  vq_gemm_wmma<<<gemmGrid(KCB), 128, 0, stream>>>(hn_h, en_h, nullptr, distp, KCB, NODES, KCB, DIM, 0);

  // VQ: argmax + gather
  vq_argmax<<<NODES, T, 0, stream>>>(distp, ind);
  vq_gather<<<ndGrid, T, 0, stream>>>(en, ind, q, q_h);

  // decoders
  vq_gemm_wmma<<<gemmGrid(DIM), 128, 0, stream>>>(q_h, Wd1t, bd1, q_edge, DIM, NODES, DIM, DIM, 0);
  vq_gemm_wmma<<<gemmGrid(DIM), 128, 0, stream>>>(q_h, Wd2t, bd2, q_node, DIM, NODES, DIM, DIM, 0);

  // commit + node reconstruction sums
  vq_mse<<<4096, T, 0, stream>>>(q, h, ND, &accs[0]);
  vq_mse<<<4096, T, 0, stream>>>(h, q_node, ND, &accs[1]);

  // adj_q min/max, then sum s^2, then sparse-edge corrections
  vq_convert_h<<<ndGrid, T, 0, stream>>>(q_edge, qe_h, ND);
  {
    int nJobs = mb * mb;                      // 35344 wave-jobs
    dim3 g((nJobs + 7) / 8);
    vq_adjq_minmax<<<g, 256, 0, stream>>>(qe_h, mnk, mxk);
    vq_minmax_final<<<1, 1, 0, stream>>>(mnk, mxk, mnf, invf);
    vq_adjq_ssq<<<g, 256, 0, stream>>>(qe_h, mnf, invf, &accs[2]);
  }
  vq_count_edges<<<(EDGES + T - 1) / T, T, 0, stream>>>(src, dst, counts);
  vq_edge_terms<<<(EDGES + 7) / 8, 256, 0, stream>>>(q_edge, src, dst, counts, mnf, invf, &accs[3]);

  // GraphConv2 on q_edge + output linear
  vq_scatter<<<EDGES, 128, 0, stream>>>(q_edge, r_out, src, dst, agg2);
  vq_scale_convert<<<ndGrid, T, 0, stream>>>(agg2, r_in, a2_h);
  vq_gemm_wmma<<<gemmGrid(DIM), 128, 0, stream>>>(a2_h, W2t, b2, h2, DIM, NODES, DIM, DIM, 1);
  vq_linear_out<<<(NODES * OUTD + T - 1) / T, T, 0, stream>>>(h2, Wl, bl, outp);

  // loss
  vq_loss_final<<<1, 1, 0, stream>>>(&accs[0], &accs[1], &accs[2], &accs[3], lossp);
}
